// Attention_26628797235884
// MI455X (gfx1250) — compile-verified, matrix operand order checked by reading, not run on hardware
//
#include <hip/hip_runtime.h>
#include <hip/hip_bf16.h>
#include <math.h>

// ---------------------------------------------------------------------------
// Problem constants (from reference)
// ---------------------------------------------------------------------------
#define BATCH 8
#define SEQ   1024
#define CH    768
#define HEADS 12
#define HDIM  64
#define N3    (3 * CH)   // 2304

typedef _Float16 half_t;
typedef __attribute__((ext_vector_type(8)))  _Float16 v8h;
typedef __attribute__((ext_vector_type(16))) _Float16 v16h;
typedef __attribute__((ext_vector_type(8)))  float    v8f;
typedef __attribute__((ext_vector_type(4)))  unsigned int u32x4;
typedef __attribute__((ext_vector_type(8)))  int      i32x8;
typedef __attribute__((ext_vector_type(4)))  int      i32x4;

#define HAVE_TDM (__has_builtin(__builtin_amdgcn_tensor_load_to_lds) && \
                  __has_builtin(__builtin_amdgcn_s_wait_tensorcnt))

// ---------------------------------------------------------------------------
// WMMA helpers
// ---------------------------------------------------------------------------
__device__ __forceinline__ v8f wmma_f16(v16h a, v16h b, v8f c) {
    // D = A(16x32 f16) * B(32x16 f16) + C(16x16 f32)
    return __builtin_amdgcn_wmma_f32_16x16x32_f16(
        /*neg_a=*/false, a, /*neg_b=*/false, b,
        /*c_mod=*/(short)0, c, /*reuse_a=*/false, /*reuse_b=*/false);
}

// Load a 16x32 (A) or 32x16 (B, from row-major transposed storage) f16
// fragment.  CDNA5 ISA layout for 16-bit A-matrices: lane t holds row
// (t & 15); element e maps to k = kbase + ((e>>3)<<4) + (t>>4)*8 + (e&7),
// i.e. two contiguous 8-half (16 byte) chunks per lane.
__device__ __forceinline__ v16h load_frag16(const half_t* __restrict__ base,
                                            int ld, int row, int kbase) {
    const int lane  = threadIdx.x & 31;
    const int khalf = lane >> 4;
    const half_t* p = base + (size_t)row * (size_t)ld + kbase + khalf * 8;
    v8h lo = *(const v8h*)(p);
    v8h hi = *(const v8h*)(p + 16);
    v16h r;
#pragma unroll
    for (int i = 0; i < 8; ++i) { r[i] = lo[i]; r[8 + i] = hi[i]; }
    return r;
}

// Same fragment pattern but reading from an LDS tile of shape [rows][32].
__device__ __forceinline__ v16h load_frag16_lds(const half_t* tile, int row) {
    const int lane  = threadIdx.x & 31;
    const int khalf = lane >> 4;
    const half_t* p = tile + row * 32 + khalf * 8;
    v8h lo = *(const v8h*)(p);
    v8h hi = *(const v8h*)(p + 16);
    v16h r;
#pragma unroll
    for (int i = 0; i < 8; ++i) { r[i] = lo[i]; r[8 + i] = hi[i]; }
    return r;
}

// ---------------------------------------------------------------------------
// Kernel 1: f32 -> f16 conversion
// ---------------------------------------------------------------------------
__global__ void cvt_f32_to_f16(const float* __restrict__ in,
                               half_t* __restrict__ out, int n) {
    int i = blockIdx.x * blockDim.x + threadIdx.x;
    if (i < n) out[i] = (half_t)in[i];
}

// ---------------------------------------------------------------------------
// Kernel 2/5: tiled WMMA GEMM with TDM-staged B tile.
//   out[M,N] = A[M,K] (f16, row-major) * W[N,K]^T (f16, row-major) + bias[N]
// Block = 128 threads = 4 waves in 2x2; block tile 128x128; per k-step the
// Tensor Data Mover stages the 128x32 f16 B slab into LDS (TENSORcnt), all
// four waves read B fragments from LDS; A fragments come straight from
// global (contiguous 16B per lane, compiler-pipelined).
// M, N multiples of 128; K multiple of 32.
// ---------------------------------------------------------------------------
__global__ __launch_bounds__(128)
void wmma_gemm(const half_t* __restrict__ A, const half_t* __restrict__ W,
               const float* __restrict__ bias,
               half_t* __restrict__ outH, float* __restrict__ outF,
               int M, int N, int K, int write_f16) {
    __shared__ half_t btile[128 * 32];       // 8 KB staged B slab

    const int lane = threadIdx.x & 31;
    const int wave = threadIdx.x >> 5;
    const int wm = wave >> 1, wn = wave & 1;
    const int row0 = blockIdx.y * 128 + wm * 64;
    const int colB = blockIdx.x * 128;       // block-level column base
    const int col0 = colB + wn * 64;
    const int mlane = lane & 15;

    v8f acc[4][4] = {};

    for (int kb = 0; kb < K; kb += 32) {
        __syncthreads();                     // WAR: previous tile fully read
#if HAVE_TDM
        if (wave == 0) {
            // --- Tensor DMA descriptor (D#): 2D tile, 2-byte elements ---
            // tile_dim0 (X) = 32 along k, tile_dim1 (Y) = 128 rows of W;
            // tensor dims/stride describe W so OOB never triggers.
            const unsigned long long gaddr =
                (unsigned long long)(uintptr_t)(W + (size_t)colB * K + kb);
            const unsigned int lds_off =
                (unsigned int)(uintptr_t)(&btile[0]);
            u32x4 g0;
            g0[0] = 1u;                                   // count=1, user mode
            g0[1] = lds_off;                              // lds_addr
            g0[2] = (unsigned int)gaddr;                  // global_addr[31:0]
            g0[3] = (unsigned int)((gaddr >> 32) & 0x01FFFFFFu)
                    | (2u << 30);                         // addr[56:32]|type=2
            i32x8 g1;
            g1[0] = 1 << 16;                              // data_size=2B
            g1[1] = (K & 0xFFFF) << 16;                   // tensor_dim0 lo16
            g1[2] = ((K >> 16) & 0xFFFF) | ((N & 0xFFFF) << 16);
            g1[3] = ((N >> 16) & 0xFFFF) | (32 << 16);    // tile_dim0=32
            g1[4] = 128;                                  // tile_dim1=128
            g1[5] = K;                                    // dim0_stride lo32
            g1[6] = 0;
            g1[7] = 0;
            i32x4 zz4 = {0, 0, 0, 0};
            i32x8 zz8 = {0, 0, 0, 0, 0, 0, 0, 0};
            // 6-arg form (clang-23 / therock-10.0 lane):
            //   (g0, g1, group2, group3, extra, cpol)
            __builtin_amdgcn_tensor_load_to_lds(g0, g1, zz4, zz4, zz8, 0);
            __builtin_amdgcn_s_wait_tensorcnt(0);
        }
#else
        // Fallback: cooperative global->LDS copy, 16B per thread per step.
        for (int idx = threadIdx.x; idx < 128 * 32 / 8; idx += 128) {
            const int r = (idx * 8) >> 5, c = (idx * 8) & 31;
            *(v8h*)&btile[r * 32 + c] =
                *(const v8h*)(W + (size_t)(colB + r) * K + kb + c);
        }
#endif
        __syncthreads();                     // RAW: tile visible to all waves

        v16h a[4], b[4];
#pragma unroll
        for (int i = 0; i < 4; ++i)
            a[i] = load_frag16(A, K, row0 + i * 16 + mlane, kb);
#pragma unroll
        for (int j = 0; j < 4; ++j)
            b[j] = load_frag16_lds(btile, wn * 64 + j * 16 + mlane);

        if (kb + 64 < K)                     // hint next A slab (prefetch_b8)
            __builtin_prefetch(A + (size_t)(row0 + mlane) * K + kb + 64, 0, 1);

#pragma unroll
        for (int i = 0; i < 4; ++i)
#pragma unroll
            for (int j = 0; j < 4; ++j)
                acc[i][j] = wmma_f16(a[i], b[j], acc[i][j]);
    }

    // C layout: lane t -> N = (t&15); VGPR r -> M = r + 8*(t>>4)
    const int rhalf = (lane >> 4) * 8;
#pragma unroll
    for (int i = 0; i < 4; ++i) {
#pragma unroll
        for (int j = 0; j < 4; ++j) {
            const int col = col0 + j * 16 + mlane;
            const float bv = bias[col];
#pragma unroll
            for (int r = 0; r < 8; ++r) {
                const int mr = row0 + i * 16 + rhalf + r;
                const float v = acc[i][j][r] + bv;
                if (write_f16) outH[(size_t)mr * N + col] = (half_t)v;
                else           outF[(size_t)mr * N + col] = v;
            }
        }
    }
}

// ---------------------------------------------------------------------------
// Kernel 3: RMSNorm + 2D RoPE + head reshape.
//  Input : qkv f16 [B*S, 3*C]  (col = which*768 + h*64 + d)
//  Output: q,k f16 [B,H,S,D];  v f16 transposed [B,H,D,S]
// ---------------------------------------------------------------------------
__global__ void rope_reshape(const half_t* __restrict__ qkv,
                             const float* __restrict__ qw,
                             const float* __restrict__ kw,
                             half_t* __restrict__ qo,
                             half_t* __restrict__ ko,
                             half_t* __restrict__ vt) {
    const int idx = blockIdx.x * blockDim.x + threadIdx.x;
    if (idx >= BATCH * SEQ * HEADS) return;
    const int h   = idx % HEADS;
    const int row = idx / HEADS;           // b*S + s
    const int bq  = row >> 10;             // batch
    const int s   = row & (SEQ - 1);
    const int y   = s >> 5;                // 2D position (h-axis)
    const int x   = s & 31;                // 2D position (w-axis)

    const half_t* src = qkv + (size_t)row * N3 + h * HDIM;

    float qv[HDIM], kv[HDIM];
    half_t vv[HDIM];
    float sq = 0.f, sk = 0.f;
#pragma unroll
    for (int c = 0; c < 8; ++c) {
        v8h tq = *(const v8h*)(src + c * 8);
        v8h tk = *(const v8h*)(src + CH + c * 8);
        v8h tv = *(const v8h*)(src + 2 * CH + c * 8);
#pragma unroll
        for (int u = 0; u < 8; ++u) {
            const float fq = (float)tq[u], fk = (float)tk[u];
            qv[c * 8 + u] = fq; kv[c * 8 + u] = fk; vv[c * 8 + u] = tv[u];
            sq += fq * fq; sk += fk * fk;
        }
    }
    const float rq = rsqrtf(sq * (1.f / HDIM) + 1e-6f);
    const float rk = rsqrtf(sk * (1.f / HDIM) + 1e-6f);

    float qr[HDIM], kr[HDIM];
#pragma unroll
    for (int j = 0; j < 32; ++j) {           // rotary pairs (2j, 2j+1)
        const int d0 = 2 * j, d1 = d0 + 1;
        const int ai = (d0 < 32) ? y : x;    // first half: h-axis, second: w
        const int jj = (d0 & 31) >> 1;
        const float fr  = __powf(10000.f, -(float)jj * (1.f / 16.f));
        const float ang = (float)ai * 0.5f * fr;
        const float cs = __cosf(ang), sn = __sinf(ang);
        const float q0 = qv[d0] * rq * qw[d0], q1 = qv[d1] * rq * qw[d1];
        const float k0 = kv[d0] * rk * kw[d0], k1 = kv[d1] * rk * kw[d1];
        qr[d0] = q0 * cs - q1 * sn;  qr[d1] = q1 * cs + q0 * sn;
        kr[d0] = k0 * cs - k1 * sn;  kr[d1] = k1 * cs + k0 * sn;
    }

    const size_t bh = (size_t)(bq * HEADS + h);
    half_t* qd = qo + (bh * SEQ + s) * HDIM;
    half_t* kd = ko + (bh * SEQ + s) * HDIM;
#pragma unroll
    for (int c = 0; c < 8; ++c) {
        v8h tq, tk;
#pragma unroll
        for (int u = 0; u < 8; ++u) {
            tq[u] = (half_t)qr[c * 8 + u];
            tk[u] = (half_t)kr[c * 8 + u];
        }
        *(v8h*)(qd + c * 8) = tq;
        *(v8h*)(kd + c * 8) = tk;
    }
    half_t* vd = vt + bh * HDIM * SEQ + s;   // [B,H,D,S]
#pragma unroll
    for (int d = 0; d < HDIM; ++d) vd[(size_t)d * SEQ] = vv[d];
}

// ---------------------------------------------------------------------------
// Kernel 4: flash attention (online softmax), f16 WMMA, f32 accumulation.
// One wave per 16-query tile; 32-key chunks; P transposed to A-fragment
// layout through padded LDS.  Output f16 [B, S, H*D].
// ---------------------------------------------------------------------------
__global__ __launch_bounds__(128)
void attn_kernel(const half_t* __restrict__ q, const half_t* __restrict__ k,
                 const half_t* __restrict__ vt, half_t* __restrict__ out) {
    __shared__ float pbuf[4][16][33];        // +1 pad: conflict-free transpose

    const int lane = threadIdx.x & 31;
    const int wave = threadIdx.x >> 5;
    const int bh = blockIdx.y;
    const int bq = bh / HEADS, h = bh % HEADS;
    const int qtile = blockIdx.x * 4 + wave;
    const int qrow0 = qtile * 16;
    const int mlane = lane & 15;
    const int half_sel = lane >> 4;

    const half_t* qb = q  + (size_t)bh * SEQ * HDIM;
    const half_t* kb = k  + (size_t)bh * SEQ * HDIM;
    const half_t* vb = vt + (size_t)bh * HDIM * SEQ;

    // Q fragments (loop invariant), pre-scaled by 1/sqrt(D) = 1/8 (exact)
    v16h aq0 = load_frag16(qb, HDIM, qrow0 + mlane, 0);
    v16h aq1 = load_frag16(qb, HDIM, qrow0 + mlane, 32);
#pragma unroll
    for (int e = 0; e < 16; ++e) {
        aq0[e] = aq0[e] * (_Float16)0.125f;
        aq1[e] = aq1[e] * (_Float16)0.125f;
    }

    float mrow[8], lrow[8];
#pragma unroll
    for (int r = 0; r < 8; ++r) { mrow[r] = -1e30f; lrow[r] = 0.f; }
    v8f acc[4] = {};

    for (int kc = 0; kc < SEQ / 32; ++kc) {
        const int base = kc * 32;

        // ---- scores: two 16x16 tiles, each K=64 via two x32 WMMAs ----
        v8f st[2];
#pragma unroll
        for (int t = 0; t < 2; ++t) {
            const int nrow = base + t * 16 + mlane;
            v16h b0 = load_frag16(kb, HDIM, nrow, 0);
            v16h b1 = load_frag16(kb, HDIM, nrow, 32);
            v8f s = {};
            s = wmma_f16(aq0, b0, s);
            s = wmma_f16(aq1, b1, s);
            st[t] = s;
        }

        // ---- online softmax (row lives across 16 lanes of a half-wave) ----
        float mnew[8], corr[8];
#pragma unroll
        for (int r = 0; r < 8; ++r) {
            float c = fmaxf(st[0][r], st[1][r]);
            c = fmaxf(c, __shfl_xor(c, 1, 32));
            c = fmaxf(c, __shfl_xor(c, 2, 32));
            c = fmaxf(c, __shfl_xor(c, 4, 32));
            c = fmaxf(c, __shfl_xor(c, 8, 32));
            mnew[r] = fmaxf(mrow[r], c);
            corr[r] = __expf(mrow[r] - mnew[r]);
            mrow[r] = mnew[r];
        }
#pragma unroll
        for (int r = 0; r < 8; ++r) {
            const float p0 = __expf(st[0][r] - mnew[r]);
            const float p1 = __expf(st[1][r] - mnew[r]);
            const int prow = r + half_sel * 8;
            pbuf[wave][prow][mlane]      = p0;
            pbuf[wave][prow][mlane + 16] = p1;
            float sum = p0 + p1;
            sum += __shfl_xor(sum, 1, 32);
            sum += __shfl_xor(sum, 2, 32);
            sum += __shfl_xor(sum, 4, 32);
            sum += __shfl_xor(sum, 8, 32);
            lrow[r] = lrow[r] * corr[r] + sum;
        }
#pragma unroll
        for (int dt = 0; dt < 4; ++dt)
#pragma unroll
            for (int r = 0; r < 8; ++r) acc[dt][r] *= corr[r];

        // ---- transpose P (C layout) -> A fragment via LDS ----
        // LDS ops complete in order within a wave (CDNA5 DScnt rules).
        v16h pf;
#pragma unroll
        for (int e = 0; e < 16; ++e) {
            const int kl = ((e >> 3) << 4) + half_sel * 8 + (e & 7);
            pf[e] = (half_t)pbuf[wave][mlane][kl];
        }

        // ---- PV: V^T rows are contiguous, 4 output d-tiles ----
#pragma unroll
        for (int dt = 0; dt < 4; ++dt) {
            v16h bv = load_frag16(vb, SEQ, dt * 16 + mlane, base);
            acc[dt] = wmma_f16(pf, bv, acc[dt]);
        }
    }

    // ---- epilogue: normalize by l, scatter heads into [B, S, H*D] ----
#pragma unroll
    for (int dt = 0; dt < 4; ++dt) {
#pragma unroll
        for (int r = 0; r < 8; ++r) {
            const int mr = qrow0 + r + half_sel * 8;
            const int d  = dt * 16 + mlane;
            const float v = acc[dt][r] / lrow[r];
            out[((size_t)(bq * SEQ + mr)) * CH + h * HDIM + d] = (half_t)v;
        }
    }
}

// ---------------------------------------------------------------------------
// Host-side launcher
// ---------------------------------------------------------------------------
extern "C" void kernel_launch(void* const* d_in, const int* in_sizes, int n_in,
                              void* d_out, int out_size, void* d_ws, size_t ws_size,
                              hipStream_t stream) {
    (void)in_sizes; (void)n_in; (void)out_size; (void)ws_size;

    const float* x      = (const float*)d_in[0];
    const float* qkv_w  = (const float*)d_in[1];
    const float* qkv_b  = (const float*)d_in[2];
    const float* q_nw   = (const float*)d_in[3];
    const float* k_nw   = (const float*)d_in[4];
    const float* proj_w = (const float*)d_in[5];
    const float* proj_b = (const float*)d_in[6];
    float* out = (float*)d_out;

    // workspace layout (f16 staging buffers)
    char* ws = (char*)d_ws;
    const size_t SZ_X16   = (size_t)BATCH * SEQ * CH * sizeof(half_t);
    const size_t SZ_QW16  = (size_t)N3 * CH * sizeof(half_t);
    const size_t SZ_PW16  = (size_t)CH * CH * sizeof(half_t);
    const size_t SZ_QKV16 = (size_t)BATCH * SEQ * N3 * sizeof(half_t);
    const size_t SZ_HEAD  = (size_t)BATCH * HEADS * SEQ * HDIM * sizeof(half_t);
    size_t o = 0;
    half_t* x16   = (half_t*)(ws + o); o += SZ_X16;
    half_t* qw16  = (half_t*)(ws + o); o += SZ_QW16;
    half_t* pw16  = (half_t*)(ws + o); o += SZ_PW16;
    half_t* qkv16 = (half_t*)(ws + o); o += SZ_QKV16;
    half_t* q16   = (half_t*)(ws + o); o += SZ_HEAD;
    half_t* k16   = (half_t*)(ws + o); o += SZ_HEAD;
    half_t* vt16  = (half_t*)(ws + o); o += SZ_HEAD;
    half_t* att16 = (half_t*)(ws + o); o += SZ_X16;

    // 1) convert inputs to f16
    {
        int n = BATCH * SEQ * CH;
        cvt_f32_to_f16<<<(n + 255) / 256, 256, 0, stream>>>(x, x16, n);
        n = N3 * CH;
        cvt_f32_to_f16<<<(n + 255) / 256, 256, 0, stream>>>(qkv_w, qw16, n);
        n = CH * CH;
        cvt_f32_to_f16<<<(n + 255) / 256, 256, 0, stream>>>(proj_w, pw16, n);
    }

    // 2) QKV GEMM: [8192,768] x [2304,768]^T + b -> f16 [8192,2304]
    {
        dim3 grid(N3 / 128, (BATCH * SEQ) / 128);
        wmma_gemm<<<grid, 128, 0, stream>>>(x16, qw16, qkv_b, qkv16, nullptr,
                                            BATCH * SEQ, N3, CH, 1);
    }

    // 3) RMSNorm + RoPE + reshape to [B,H,S,D] (q,k) and [B,H,D,S] (v)
    {
        int n = BATCH * SEQ * HEADS;
        rope_reshape<<<(n + 255) / 256, 256, 0, stream>>>(qkv16, q_nw, k_nw,
                                                          q16, k16, vt16);
    }

    // 4) flash attention -> f16 [B, S, H*D]
    {
        dim3 grid((SEQ / 16) / 4, BATCH * HEADS);
        attn_kernel<<<grid, 128, 0, stream>>>(q16, k16, vt16, att16);
    }

    // 5) projection GEMM: [8192,768] x [768,768]^T + b -> f32 d_out
    {
        dim3 grid(CH / 128, (BATCH * SEQ) / 128);
        wmma_gemm<<<grid, 128, 0, stream>>>(att16, pw16, proj_b, nullptr, out,
                                            BATCH * SEQ, CH, CH, 0);
    }
}
